// PerTokenFFN_77876347011612
// MI455X (gfx1250) — compile-verified, matrix operand order checked by reading, not run on hardware
//
#include <hip/hip_runtime.h>
#include <hip/hip_bf16.h>
#include <math.h>

// ---- problem constants (match reference) ----
#define T_ 32
#define D_ 256
#define H_ 1024
#define B_ 2048

// ---- tiling ----
#define BM 128      // rows of x per workgroup
#define HC 64       // H chunk per iteration
#define NWAVE 8     // 256 threads / wave32

// LDS layout (bytes):
//   xA : [BM][D_]   bf16 = 65536
//   w1s: [HC][D_]   bf16 = 32768   (B^T of layer-1 chunk: row = h, col = d)
//   w2s: [D_][HC]   bf16 = 32768   (B^T of layer-2:       row = d, col = h)
//   hs : [NWAVE][16][HC] bf16 = 16384 (per-wave gelu(h) staging for A-frags)
#define LDS_XA   0
#define LDS_W1S  65536
#define LDS_W2S  (65536 + 32768)
#define LDS_HS   (65536 + 32768 + 32768)
#define LDS_BYTES (65536 + 32768 + 32768 + 16384)

typedef __attribute__((ext_vector_type(16))) __bf16 v16bf;
typedef __attribute__((ext_vector_type(8)))  float  v8f;
typedef __attribute__((ext_vector_type(4)))  unsigned int u32x4;
typedef __attribute__((ext_vector_type(2)))  unsigned int u32x2;

__device__ __forceinline__ unsigned pack_bf16x2(float a, float b) {
    unsigned short ua = __builtin_bit_cast(unsigned short, (__bf16)a);
    unsigned short ub = __builtin_bit_cast(unsigned short, (__bf16)b);
    return (unsigned)ua | ((unsigned)ub << 16);
}

// Branch-free exact-erf GELU (Abramowitz & Stegun 7.1.26, |err| < 1.5e-7 --
// below the bf16 rounding applied to h anyway). Uses v_rcp_f32 + v_exp_f32,
// no EXEC divergence, unlike ocml erff.
__device__ __forceinline__ float gelu_erf(float v) {
    float z = fabsf(v) * 0.70710678118654752f;     // |v|/sqrt(2)
    float t = __builtin_amdgcn_rcpf(fmaf(0.3275911f, z, 1.0f));
    float p = t * fmaf(t, fmaf(t, fmaf(t, fmaf(t, 1.061405429f, -1.453152027f),
                                       1.421413741f), -0.284496736f), 0.254829592f);
    float e = __expf(-z * z);
    float erfv = copysignf(fmaf(-p, e, 1.0f), v);  // erf(v/sqrt(2))
    return 0.5f * v * (1.0f + erfv);
}

// Build one 16x16x32 bf16 operand fragment (8 VGPRs) from a row base in LDS.
// Per the CDNA5 ISA A-layout: this lane's data is two contiguous 16-byte runs
// at element offsets koff and koff+16 within its row.
__device__ __forceinline__ v16bf ld_frag(const __bf16* rowptr, int koff) {
    union { v16bf v; u32x4 q[2]; } u;
    u.q[0] = *(const u32x4*)(rowptr + koff);
    u.q[1] = *(const u32x4*)(rowptr + koff + 16);
    return u.v;
}

__device__ __forceinline__ v8f wmma_bf16(v16bf a, v16bf b, v8f c) {
    return __builtin_amdgcn_wmma_f32_16x16x32_bf16(false, a, false, b,
                                                   (short)0, c, false, false);
}

__global__ void __launch_bounds__(256)
PerTokenFFN_wmma_kernel(const float* __restrict__ x,
                        const float* __restrict__ W1,
                        const float* __restrict__ b1,
                        const float* __restrict__ W2,
                        const float* __restrict__ b2,
                        float* __restrict__ y)
{
    extern __shared__ char smem[];
    __bf16* xA    = (__bf16*)(smem + LDS_XA);
    __bf16* w1s   = (__bf16*)(smem + LDS_W1S);
    __bf16* w2s   = (__bf16*)(smem + LDS_W2S);
    __bf16* hsAll = (__bf16*)(smem + LDS_HS);

    const int tid   = threadIdx.x;
    const int lane  = tid & 31;
    const int wave  = tid >> 5;
    const int half  = lane >> 4;   // which half-wave (selects K sub-range)
    const int lhalf = lane & 15;   // M (A-frag) / N (B-frag) index
    const int tkn   = blockIdx.y;
    const int b0    = blockIdx.x * BM;
    const int m0    = wave * 16;

    __bf16* hs = hsAll + wave * (16 * HC);

    // ---------------- stage x tile (f32 -> bf16) ----------------
    {
        const float* xbase = x + (size_t)b0 * (T_ * D_) + (size_t)tkn * D_;
        #pragma unroll
        for (int i = 0; i < 32; ++i) {
            int idx = tid + i * 256;
            int row = idx >> 6;        // 64 float4 per row
            int c4  = idx & 63;
            const float4 v = *(const float4*)(xbase + (size_t)row * (T_ * D_) + c4 * 4);
            u32x2 p;
            p.x = pack_bf16x2(v.x, v.y);
            p.y = pack_bf16x2(v.z, v.w);
            *(u32x2*)(xA + row * D_ + c4 * 4) = p;
        }
    }

    v8f yacc[16];
    #pragma unroll
    for (int n = 0; n < 16; ++n) yacc[n] = (v8f){0.f,0.f,0.f,0.f,0.f,0.f,0.f,0.f};

    for (int hc = 0; hc < H_; hc += HC) {
        __syncthreads();  // previous chunk's fragment reads complete

        // ---- stage W1 chunk transposed: w1s[h][d], thread <-> d row ----
        {
            const float* wrow = W1 + ((size_t)tkn * D_ + tid) * H_ + hc;
            #pragma unroll
            for (int j = 0; j < HC; j += 4) {
                float4 v = *(const float4*)(wrow + j);
                w1s[(j + 0) * D_ + tid] = (__bf16)v.x;
                w1s[(j + 1) * D_ + tid] = (__bf16)v.y;
                w1s[(j + 2) * D_ + tid] = (__bf16)v.z;
                w1s[(j + 3) * D_ + tid] = (__bf16)v.w;
            }
        }
        // ---- stage W2 chunk transposed: w2s[d][h], thread <-> (h, d-seg) ----
        {
            int hh = tid >> 2;
            int d0 = (tid & 3) * 64;
            const float* wrow = W2 + ((size_t)tkn * H_ + hc + hh) * D_ + d0;
            #pragma unroll
            for (int j = 0; j < 64; j += 4) {
                float4 v = *(const float4*)(wrow + j);
                w2s[(d0 + j + 0) * HC + hh] = (__bf16)v.x;
                w2s[(d0 + j + 1) * HC + hh] = (__bf16)v.y;
                w2s[(d0 + j + 2) * HC + hh] = (__bf16)v.z;
                w2s[(d0 + j + 3) * HC + hh] = (__bf16)v.w;
            }
        }
        __syncthreads();

        // ---- GEMM1: h[16 x 64] = xA[16 x 256] * W1chunk[256 x 64] ----
        // Software-pipelined: fragments for k-step s+1 are loaded while the
        // WMMAs of k-step s execute, so ds_loads overlap matrix issue.
        v8f ht[4];
        #pragma unroll
        for (int n = 0; n < 4; ++n) ht[n] = (v8f){0.f,0.f,0.f,0.f,0.f,0.f,0.f,0.f};

        {
            const __bf16* aRow = xA + (m0 + lhalf) * D_;
            v16bf a_cur = ld_frag(aRow, half * 8);
            v16bf b_cur[4];
            #pragma unroll
            for (int n = 0; n < 4; ++n)
                b_cur[n] = ld_frag(w1s + (n * 16 + lhalf) * D_, half * 8);

            #pragma unroll
            for (int ks = 0; ks < 8; ++ks) {
                v16bf a_nxt;
                v16bf b_nxt[4];
                if (ks < 7) {
                    int koff = (ks + 1) * 32 + half * 8;
                    a_nxt = ld_frag(aRow, koff);
                    #pragma unroll
                    for (int n = 0; n < 4; ++n)
                        b_nxt[n] = ld_frag(w1s + (n * 16 + lhalf) * D_, koff);
                }
                #pragma unroll
                for (int n = 0; n < 4; ++n)
                    ht[n] = wmma_bf16(a_cur, b_cur[n], ht[n]);
                if (ks < 7) {
                    a_cur = a_nxt;
                    #pragma unroll
                    for (int n = 0; n < 4; ++n) b_cur[n] = b_nxt[n];
                }
            }
        }

        // ---- bias + exact GELU, round to bf16, scatter into A-layout LDS ----
        #pragma unroll
        for (int n = 0; n < 4; ++n) {
            float bias = b1[(size_t)tkn * H_ + hc + n * 16 + lhalf];
            #pragma unroll
            for (int r = 0; r < 8; ++r) {
                float v = ht[n][r] + bias;          // C/D: row r+8*half, col lhalf
                hs[(r + 8 * half) * HC + n * 16 + lhalf] = (__bf16)gelu_erf(v);
            }
        }

        // ---- GEMM2: yacc[16 x 256] += gelu(h)[16 x 64] * W2chunk[64 x 256] ----
        // 8 pipeline steps: step s = (kk-half s>>2, ntile-group s&3), 4 WMMAs each.
        {
            const __bf16* hRow = hs + lhalf * HC;
            v16bf a2[2];
            a2[0] = ld_frag(hRow, half * 8);
            a2[1] = ld_frag(hRow, 32 + half * 8);

            v16bf b_cur[4];
            #pragma unroll
            for (int j = 0; j < 4; ++j)
                b_cur[j] = ld_frag(w2s + (j * 16 + lhalf) * HC, half * 8);

            #pragma unroll
            for (int s = 0; s < 8; ++s) {
                v16bf b_nxt[4];
                if (s < 7) {
                    int sn  = s + 1;
                    int kkn = (sn >> 2) * 32;
                    int gn  = sn & 3;
                    #pragma unroll
                    for (int j = 0; j < 4; ++j)
                        b_nxt[j] = ld_frag(w2s + ((gn * 4 + j) * 16 + lhalf) * HC,
                                           kkn + half * 8);
                }
                int g = s & 3;
                #pragma unroll
                for (int j = 0; j < 4; ++j)
                    yacc[g * 4 + j] = wmma_bf16(a2[s >> 2], b_cur[j], yacc[g * 4 + j]);
                if (s < 7) {
                    #pragma unroll
                    for (int j = 0; j < 4; ++j) b_cur[j] = b_nxt[j];
                }
            }
        }
    }

    // ---------------- epilogue: + b2, store f32 ----------------
    #pragma unroll
    for (int n = 0; n < 16; ++n) {
        int col    = n * 16 + lhalf;
        float bias = b2[(size_t)tkn * D_ + col];
        #pragma unroll
        for (int r = 0; r < 8; ++r) {
            int row = m0 + r + 8 * half;
            y[(size_t)(b0 + row) * (T_ * D_) + (size_t)tkn * D_ + col] = yacc[n][r] + bias;
        }
    }
}

extern "C" void kernel_launch(void* const* d_in, const int* in_sizes, int n_in,
                              void* d_out, int out_size, void* d_ws, size_t ws_size,
                              hipStream_t stream) {
    (void)in_sizes; (void)n_in; (void)out_size; (void)d_ws; (void)ws_size;
    const float* x  = (const float*)d_in[0];
    const float* W1 = (const float*)d_in[1];
    const float* b1 = (const float*)d_in[2];
    const float* W2 = (const float*)d_in[3];
    const float* b2 = (const float*)d_in[4];
    float* y = (float*)d_out;

    dim3 grid(B_ / BM, T_);   // 16 x 32 = 512 workgroups
    dim3 block(256);          // 8 wave32
    PerTokenFFN_wmma_kernel<<<grid, block, LDS_BYTES, stream>>>(x, W1, b1, W2, b2, y);
}
